// TransformerBlock_24240795419483
// MI455X (gfx1250) — compile-verified, matrix operand order checked by reading, not run on hardware
//
#include <hip/hip_runtime.h>
#include <hip/hip_bf16.h>
#include <math.h>

// ---------------------------------------------------------------------------
// Transformer block for MI455X (gfx1250, wave32, WMMA + TDM).
// GEMMs: bf16 in, fp32 accumulate via v_wmma_f32_16x16x32_bf16.
// A-tiles staged into LDS by the Tensor Data Mover (TENSORcnt-tracked),
// double-buffered so DMA overlaps WMMA. Attention is flash-style.
// ---------------------------------------------------------------------------

typedef __attribute__((ext_vector_type(16))) __bf16 v16bf;
typedef __attribute__((ext_vector_type(8)))  __bf16 v8bf;
typedef __attribute__((ext_vector_type(8)))  float  v8f;
typedef unsigned int v4u __attribute__((ext_vector_type(4)));
typedef int          v4i __attribute__((ext_vector_type(4)));
typedef int          v8i __attribute__((ext_vector_type(8)));

#define WMMA_BF16(a, b, c) \
  __builtin_amdgcn_wmma_f32_16x16x32_bf16(false, (a), false, (b), (short)0, (c), false, false)

#ifndef __has_builtin
#define __has_builtin(x) 0
#endif

#if __has_builtin(__builtin_amdgcn_tensor_load_to_lds) && __has_builtin(__builtin_amdgcn_s_wait_tensorcnt)
#define USE_TDM 1
#else
#define USE_TDM 0
#endif

// therock-10.0 headers (amdgpu-toolchain) ship the TDM header and the 6-arg
// builtin; ROCm 7.2 uses the 5-arg form.
#if defined(__has_include)
#if __has_include(<hip/amd_detail/amd_gfx1250_TDM.h>)
#define TDM_6ARG 1
#endif
#endif

// Problem constants (reference: B=4, S=2048, D=512, H=8, dh=64)
#define PB 4
#define PS 2048
#define PD 512
#define PH 8
#define PDH 64

enum { MODE_F32 = 0, MODE_BF16 = 1, MODE_HEADS = 2, MODE_HEADS_T = 3, MODE_GELU = 4 };

// ---------------------------------------------------------------------------
// f32 -> bf16 conversion
// ---------------------------------------------------------------------------
__global__ void f32_to_bf16(const float* __restrict__ in, __bf16* __restrict__ out, int n) {
  int i = blockIdx.x * 256 + threadIdx.x;
  if (i < n) out[i] = (__bf16)in[i];
}

#if USE_TDM
// ---------------------------------------------------------------------------
// Issue a TDM 2D tile load: tile_h rows x tile_w bf16 elems, global row
// stride row_stride elems, into LDS at byte offset ldsOff (row-major).
// D# packing per CDNA5 ISA 8.3/8.4.
// ---------------------------------------------------------------------------
__device__ __forceinline__ void tdm_load_2d(unsigned int ldsOff, const __bf16* gptr,
                                            unsigned int width, unsigned int rows,
                                            unsigned int tile_w, unsigned int tile_h,
                                            unsigned int row_stride) {
  unsigned long long ga = (unsigned long long)(uintptr_t)gptr;
  v4u g0;
  g0[0] = 1u;                                            // count=1 user descriptor
  g0[1] = ldsOff;                                        // lds_addr (bytes)
  g0[2] = (unsigned int)ga;                              // global_addr[31:0]
  g0[3] = (unsigned int)((ga >> 32) & 0x01FFFFFFu)       // global_addr[56:32]
          | (2u << 30);                                  // type = 2 ("image")
  v8i g1;
  g1[0] = 0x00010000;                                    // data_size=1 (2B), mask=0
  g1[1] = (int)((width & 0xFFFFu) << 16);                // tensor_dim0[15:0]
  g1[2] = (int)((width >> 16) | ((rows & 0xFFFFu) << 16));   // dim0 hi | dim1 lo
  g1[3] = (int)(((rows >> 16) & 0xFFFFu) | (tile_w << 16));  // dim1 hi | tile_dim0
  g1[4] = (int)(tile_h & 0xFFFFu);                       // tile_dim1 (tile_dim2=0)
  g1[5] = (int)row_stride;                               // tensor_dim0_stride lo32
  g1[6] = 0;                                             // stride hi | dim1_stride lo
  g1[7] = 0;
  v4i z4 = {0, 0, 0, 0};
#if defined(TDM_6ARG)
  v8i z8 = {0, 0, 0, 0, 0, 0, 0, 0};
  __builtin_amdgcn_tensor_load_to_lds(g0, g1, z4, z4, z8, 0);
#else
  __builtin_amdgcn_tensor_load_to_lds(g0, g1, z4, z4, 0);
#endif
}
#endif  // USE_TDM

// ---------------------------------------------------------------------------
// Tiled WMMA GEMM: C[M,N] = A[M,K](bf16) @ B[K,N](bf16) + bias, epilogue modes.
// Block tile 128x64, 8 waves in 4x2; each wave owns 32x32 (4 WMMAs/K-step).
// Double-buffered LDS: TDM loads A tile i+1 and global loads of B tile i+1
// are issued before computing tile i; drained after the WMMAs.
// ---------------------------------------------------------------------------
__global__ __launch_bounds__(256)
void gemm_bf16(const __bf16* __restrict__ A, const __bf16* __restrict__ Bw,
               const float* __restrict__ bias,
               float* __restrict__ outF, __bf16* __restrict__ outB,
               int M, int N, int K, float scale, int mode)
{
  __shared__ __bf16 As[2][128 * 32];  // [m][k] row-major (2 x 8KB)
  __shared__ __bf16 Bs[2][64 * 32];   // transposed [n][k]  (2 x 4KB)

  const int nTiles = N >> 6;
  const int mTile  = blockIdx.x / nTiles;
  const int nTile  = blockIdx.x % nTiles;
  const int t    = threadIdx.x;
  const int lane = t & 31;
  const int wave = t >> 5;
  const int wm   = wave >> 1;         // 0..3 -> 32-row band
  const int wn   = wave & 1;          // 0..1 -> 32-col band
  const int l16  = lane & 15;
  const int off  = (lane & 16) ? 8 : 0;   // K-half select per ISA A/B layout

  v8f a00 = {}, a01 = {}, a10 = {}, a11 = {};

  // Staging assignments
  const int aM = t >> 1;              // 0..127
  const int aK = (t & 1) << 4;        // 0,16
  const int bK = t >> 3;              // 0..31
  const int bN = (t & 7) << 3;        // 0..56

  const __bf16* aRow = A + (size_t)(mTile * 128 + aM) * K + aK;     // fallback path
  const __bf16* aTdm = A + (size_t)(mTile * 128) * K;               // TDM path
  const __bf16* bRow = Bw + (size_t)bK * N + nTile * 64 + bN;
#if USE_TDM
  const unsigned int asOff0 = (unsigned int)(uintptr_t)&As[0][0];
  const unsigned int asOff1 = (unsigned int)(uintptr_t)&As[1][0];
#endif

  const int nK = K >> 5;              // K-steps of 32

  // ---- prologue: stage tile 0 into buffer 0 ----
#if USE_TDM
  if (wave == 0) tdm_load_2d(asOff0, aTdm, (unsigned)K, (unsigned)M, 32u, 128u, (unsigned)K);
  aTdm += 32;
#else
  {
    v8bf a0 = *reinterpret_cast<const v8bf*>(aRow);
    v8bf a1 = *reinterpret_cast<const v8bf*>(aRow + 8);
    aRow += 32;
    *reinterpret_cast<v8bf*>(&As[0][aM * 32 + aK]) = a0;
    *reinterpret_cast<v8bf*>(&As[0][aM * 32 + aK + 8]) = a1;
  }
#endif
  {
    v8bf bv = *reinterpret_cast<const v8bf*>(bRow);
    bRow += (size_t)32 * N;
#pragma unroll
    for (int j = 0; j < 8; ++j) Bs[0][(bN + j) * 32 + bK] = bv[j];
  }
#if USE_TDM
  if (wave == 0) __builtin_amdgcn_s_wait_tensorcnt(0);
#endif
  __syncthreads();

  // ---- pipelined main loop ----
  for (int it = 0; it < nK; ++it) {
    const int cur = it & 1;
    const int nxt = cur ^ 1;
    const bool more = (it + 1) < nK;

    // Issue next tile's memory ops first (overlap with WMMA below)
    v8bf aN0, aN1, bNreg;
    if (more) {
#if USE_TDM
      if (wave == 0)
        tdm_load_2d(nxt ? asOff1 : asOff0, aTdm, (unsigned)K, (unsigned)M,
                    32u, 128u, (unsigned)K);
      aTdm += 32;
#else
      aN0 = *reinterpret_cast<const v8bf*>(aRow);
      aN1 = *reinterpret_cast<const v8bf*>(aRow + 8);
      aRow += 32;
#endif
      bNreg = *reinterpret_cast<const v8bf*>(bRow);
      __builtin_prefetch((const void*)(bRow + (size_t)32 * N), 0, 3);
      bRow += (size_t)32 * N;
    }

    // Compute on current buffers.
    // Fragments: lane<16 holds K {0..7,16..23}, lane>=16 holds {8..15,24..31}
    const __bf16* asBase = &As[cur][0];
    const __bf16* bsBase = &Bs[cur][0];
    v16bf af[2], bfv[2];
#pragma unroll
    for (int mi = 0; mi < 2; ++mi) {
      const __bf16* rp = asBase + (wm * 32 + mi * 16 + l16) * 32;
      v8bf lo = *reinterpret_cast<const v8bf*>(rp + off);
      v8bf hi = *reinterpret_cast<const v8bf*>(rp + 16 + off);
#pragma unroll
      for (int j = 0; j < 8; ++j) { af[mi][j] = lo[j]; af[mi][8 + j] = hi[j]; }
    }
#pragma unroll
    for (int ni = 0; ni < 2; ++ni) {
      const __bf16* rp = bsBase + (wn * 32 + ni * 16 + l16) * 32;
      v8bf lo = *reinterpret_cast<const v8bf*>(rp + off);
      v8bf hi = *reinterpret_cast<const v8bf*>(rp + 16 + off);
#pragma unroll
      for (int j = 0; j < 8; ++j) { bfv[ni][j] = lo[j]; bfv[ni][8 + j] = hi[j]; }
    }
    a00 = WMMA_BF16(af[0], bfv[0], a00);
    a01 = WMMA_BF16(af[0], bfv[1], a01);
    a10 = WMMA_BF16(af[1], bfv[0], a10);
    a11 = WMMA_BF16(af[1], bfv[1], a11);

    // Drain next-tile staging into the other buffer (safe: all waves finished
    // reading it before the barrier that ended the previous iteration).
    if (more) {
#if !USE_TDM
      *reinterpret_cast<v8bf*>(&As[nxt][aM * 32 + aK]) = aN0;
      *reinterpret_cast<v8bf*>(&As[nxt][aM * 32 + aK + 8]) = aN1;
#endif
#pragma unroll
      for (int j = 0; j < 8; ++j) Bs[nxt][(bN + j) * 32 + bK] = bNreg[j];
#if USE_TDM
      if (wave == 0) __builtin_amdgcn_s_wait_tensorcnt(0);
#endif
    }
    __syncthreads();
  }

  // Epilogue. C layout: VGPR i -> M = i (lanes 0-15) / i+8 (lanes 16-31), N = lane%16.
  const int colBase = nTile * 64 + wn * 32 + l16;
  const float bias0 = bias[colBase];
  const float bias1 = bias[colBase + 16];
  const int rowBase = mTile * 128 + wm * 32 + ((lane & 16) ? 8 : 0);
#pragma unroll
  for (int mi = 0; mi < 2; ++mi) {
#pragma unroll
    for (int ni = 0; ni < 2; ++ni) {
      const v8f acc = mi ? (ni ? a11 : a10) : (ni ? a01 : a00);
      const float bb = ni ? bias1 : bias0;
      const int col = colBase + ni * 16;
#pragma unroll
      for (int i = 0; i < 8; ++i) {
        const int row = rowBase + mi * 16 + i;
        float val = acc[i] + bb;
        if (mode == MODE_F32) {
          outF[(size_t)row * N + col] = val;
        } else if (mode == MODE_BF16) {
          outB[(size_t)row * N + col] = (__bf16)val;
        } else if (mode == MODE_GELU) {
          val = 0.5f * val * (1.0f + erff(val * 0.70710678118f));   // exact GELU
          outB[(size_t)row * N + col] = (__bf16)val;
        } else {
          // head-split: row = b*S + s ; col = h*dh + d
          const int b  = row >> 11;        // /PS
          const int s  = row & (PS - 1);
          const int hh = col >> 6;         // /PDH
          const int dd = col & (PDH - 1);
          val *= scale;
          if (mode == MODE_HEADS)          // (B,H,S,dh)
            outB[(((size_t)b * PH + hh) * PS + s) * PDH + dd] = (__bf16)val;
          else                             // (B,H,dh,S)  (transposed V)
            outB[(((size_t)b * PH + hh) * PDH + dd) * PS + s] = (__bf16)val;
        }
      }
    }
  }
}

// ---------------------------------------------------------------------------
// Flash attention: one wave per (b,h, 16-query tile). KV chunks of 32 keys:
// 4 WMMAs for Q.K^T, online softmax (shfl_xor within 16-lane halves),
// P relaid through per-wave LDS into an A-fragment, 4 WMMAs for P.V.
// ---------------------------------------------------------------------------
__global__ __launch_bounds__(256)
void flash_attn(const __bf16* __restrict__ q, const __bf16* __restrict__ k,
                const __bf16* __restrict__ vt, __bf16* __restrict__ ctx)
{
  __shared__ __bf16 Plds[8 * 16 * 32];           // 1KB per wave
  const int lane = threadIdx.x & 31;
  const int wave = threadIdx.x >> 5;
  const int task = blockIdx.x * 8 + wave;        // B*H*(S/16) = 4096 tasks
  const int qt = task & ((PS / 16) - 1);         // query tile
  const int bh = task >> 7;                      // (b*H + h)
  const int l16 = lane & 15;
  const int off = (lane & 16) ? 8 : 0;

  const __bf16* qh = q  + (size_t)bh * PS * PDH;
  const __bf16* kh = k  + (size_t)bh * PS * PDH;
  const __bf16* vh = vt + (size_t)bh * PDH * PS; // (dh, S)
  const int q0 = qt * 16;

  // Q fragments: 16x64 -> two k=32 A-fragments; contiguous b128 loads
  v16bf qf[2];
#pragma unroll
  for (int kk = 0; kk < 2; ++kk) {
    const __bf16* rp = qh + (size_t)(q0 + l16) * PDH + kk * 32;
    v8bf lo = *reinterpret_cast<const v8bf*>(rp + off);
    v8bf hi = *reinterpret_cast<const v8bf*>(rp + 16 + off);
#pragma unroll
    for (int j = 0; j < 8; ++j) { qf[kk][j] = lo[j]; qf[kk][8 + j] = hi[j]; }
  }

  v8f o0 = {}, o1 = {}, o2 = {}, o3 = {};
  float mrow[8], lrow[8];
#pragma unroll
  for (int i = 0; i < 8; ++i) { mrow[i] = -3.0e38f; lrow[i] = 0.0f; }

  __bf16* pbase = &Plds[wave * 16 * 32];

  for (int kv = 0; kv < PS; kv += 32) {
    // Prefetch next chunk's K rows while we compute this one
    __builtin_prefetch((const void*)(kh + (size_t)(kv + 32 + l16) * PDH), 0, 3);
    // S tile: 16 queries x 32 keys, accumulate over dh=64 (two k=32 steps)
    v8f s0 = {}, s1 = {};
#pragma unroll
    for (int kk = 0; kk < 2; ++kk) {
#pragma unroll
      for (int ng = 0; ng < 2; ++ng) {
        const __bf16* rp = kh + (size_t)(kv + ng * 16 + l16) * PDH + kk * 32;
        v8bf lo = *reinterpret_cast<const v8bf*>(rp + off);
        v8bf hi = *reinterpret_cast<const v8bf*>(rp + 16 + off);
        v16bf kf;
#pragma unroll
        for (int j = 0; j < 8; ++j) { kf[j] = lo[j]; kf[8 + j] = hi[j]; }
        if (ng == 0) s0 = WMMA_BF16(qf[kk], kf, s0);
        else         s1 = WMMA_BF16(qf[kk], kf, s1);
      }
    }
    // Online softmax; each accumulator row lives in one 16-lane half.
#pragma unroll
    for (int i = 0; i < 8; ++i) {
      float mx = fmaxf(s0[i], s1[i]);
      mx = fmaxf(mx, __shfl_xor(mx, 1, 32));
      mx = fmaxf(mx, __shfl_xor(mx, 2, 32));
      mx = fmaxf(mx, __shfl_xor(mx, 4, 32));
      mx = fmaxf(mx, __shfl_xor(mx, 8, 32));
      const float newm  = fmaxf(mrow[i], mx);
      const float alpha = __expf(mrow[i] - newm);
      const float p0 = __expf(s0[i] - newm);
      const float p1 = __expf(s1[i] - newm);
      float rs = p0 + p1;
      rs += __shfl_xor(rs, 1, 32);
      rs += __shfl_xor(rs, 2, 32);
      rs += __shfl_xor(rs, 4, 32);
      rs += __shfl_xor(rs, 8, 32);
      lrow[i] = lrow[i] * alpha + rs;
      mrow[i] = newm;
      o0[i] *= alpha; o1[i] *= alpha; o2[i] *= alpha; o3[i] *= alpha;
      const int r = i + ((lane & 16) ? 8 : 0);
      pbase[r * 32 + l16]      = (__bf16)p0;
      pbase[r * 32 + 16 + l16] = (__bf16)p1;
    }
    // Re-read P as bf16 A-fragment (LDS ops from same wave are in-order)
    v16bf pf;
    {
      const __bf16* rp = pbase + l16 * 32;
      v8bf lo = *reinterpret_cast<const v8bf*>(rp + off);
      v8bf hi = *reinterpret_cast<const v8bf*>(rp + 16 + off);
#pragma unroll
      for (int j = 0; j < 8; ++j) { pf[j] = lo[j]; pf[8 + j] = hi[j]; }
    }
    // O += P @ V ; V stored transposed so B-fragments load contiguously
#pragma unroll
    for (int c = 0; c < 4; ++c) {
      const __bf16* rp = vh + (size_t)(c * 16 + l16) * PS + kv;
      v8bf lo = *reinterpret_cast<const v8bf*>(rp + off);
      v8bf hi = *reinterpret_cast<const v8bf*>(rp + 16 + off);
      v16bf vf;
#pragma unroll
      for (int j = 0; j < 8; ++j) { vf[j] = lo[j]; vf[8 + j] = hi[j]; }
      if (c == 0)      o0 = WMMA_BF16(pf, vf, o0);
      else if (c == 1) o1 = WMMA_BF16(pf, vf, o1);
      else if (c == 2) o2 = WMMA_BF16(pf, vf, o2);
      else             o3 = WMMA_BF16(pf, vf, o3);
    }
  }

  // Normalize and write ctx back to (B,S,D) bf16 (merged heads)
  const int b = bh >> 3, hh = bh & 7;
#pragma unroll
  for (int i = 0; i < 8; ++i) {
    const float inv = 1.0f / lrow[i];
    const int r = i + ((lane & 16) ? 8 : 0);
    const size_t rowbase = ((size_t)b * PS + q0 + r) * PD + hh * PDH;
    ctx[rowbase + 0 * 16 + l16] = (__bf16)(o0[i] * inv);
    ctx[rowbase + 1 * 16 + l16] = (__bf16)(o1[i] * inv);
    ctx[rowbase + 2 * 16 + l16] = (__bf16)(o2[i] * inv);
    ctx[rowbase + 3 * 16 + l16] = (__bf16)(o3[i] * inv);
  }
}

// ---------------------------------------------------------------------------
// LayerNorm(a + res) * g + be ; writes f32 (residual chain) and optional bf16.
// One block per row, D=512, 2 elements/thread.
// ---------------------------------------------------------------------------
__global__ __launch_bounds__(256)
void ln_residual(const float* __restrict__ a, const float* __restrict__ res,
                 const float* __restrict__ g, const float* __restrict__ be,
                 float* __restrict__ outF, __bf16* __restrict__ outB)
{
  __shared__ float red[256];
  const int row = blockIdx.x;
  const int t = threadIdx.x;
  const size_t base = (size_t)row * PD;
  float x0 = a[base + t]       + res[base + t];
  float x1 = a[base + 256 + t] + res[base + 256 + t];
  red[t] = x0 + x1;
  __syncthreads();
  for (int w = 128; w > 0; w >>= 1) { if (t < w) red[t] += red[t + w]; __syncthreads(); }
  const float mean = red[0] * (1.0f / PD);
  __syncthreads();
  const float d0 = x0 - mean, d1 = x1 - mean;
  red[t] = d0 * d0 + d1 * d1;
  __syncthreads();
  for (int w = 128; w > 0; w >>= 1) { if (t < w) red[t] += red[t + w]; __syncthreads(); }
  const float rstd = rsqrtf(red[0] * (1.0f / PD) + 1e-12f);
  const float y0 = d0 * rstd * g[t] + be[t];
  const float y1 = d1 * rstd * g[256 + t] + be[256 + t];
  outF[base + t] = y0;
  outF[base + 256 + t] = y1;
  if (outB) { outB[base + t] = (__bf16)y0; outB[base + 256 + t] = (__bf16)y1; }
}

// ---------------------------------------------------------------------------
// Host-side orchestration
// ---------------------------------------------------------------------------
extern "C" void kernel_launch(void* const* d_in, const int* in_sizes, int n_in,
                              void* d_out, int out_size, void* d_ws, size_t ws_size,
                              hipStream_t stream) {
  (void)in_sizes; (void)n_in; (void)out_size; (void)ws_size;
  const float* x   = (const float*)d_in[0];
  const float* wq  = (const float*)d_in[1];  const float* bq  = (const float*)d_in[2];
  const float* wk  = (const float*)d_in[3];  const float* bk  = (const float*)d_in[4];
  const float* wv  = (const float*)d_in[5];  const float* bv  = (const float*)d_in[6];
  const float* wo  = (const float*)d_in[7];  const float* bo  = (const float*)d_in[8];
  const float* w1  = (const float*)d_in[9];  const float* b1  = (const float*)d_in[10];
  const float* w2  = (const float*)d_in[11]; const float* b2  = (const float*)d_in[12];
  const float* g1  = (const float*)d_in[13]; const float* be1 = (const float*)d_in[14];
  const float* g2  = (const float*)d_in[15]; const float* be2 = (const float*)d_in[16];
  float* out = (float*)d_out;

  const size_t ND = (size_t)PB * PS * PD;        // 4,194,304 elements
  const size_t WD = (size_t)PD * PD;             // 262,144

  char* ws = (char*)d_ws;
  size_t off = 0;
  auto alloc = [&](size_t bytes) -> void* {
    void* p = ws + off;
    off += (bytes + 255) & ~(size_t)255;
    return p;
  };
  __bf16* xb   = (__bf16*)alloc(ND * 2);
  __bf16* wqb  = (__bf16*)alloc(WD * 2);
  __bf16* wkb  = (__bf16*)alloc(WD * 2);
  __bf16* wvb  = (__bf16*)alloc(WD * 2);
  __bf16* wob  = (__bf16*)alloc(WD * 2);
  __bf16* w1b  = (__bf16*)alloc(WD * 2);
  __bf16* w2b  = (__bf16*)alloc(WD * 2);
  __bf16* qb   = (__bf16*)alloc(ND * 2);   // (B,H,S,dh), pre-scaled by 1/sqrt(dh)
  __bf16* kb   = (__bf16*)alloc(ND * 2);   // (B,H,S,dh)
  __bf16* vtb  = (__bf16*)alloc(ND * 2);   // (B,H,dh,S)
  __bf16* ctxb = (__bf16*)alloc(ND * 2);   // (B,S,D)
  float*  sa   = (float*) alloc(ND * 4);
  float*  hf   = (float*) alloc(ND * 4);
  __bf16* hb   = (__bf16*)alloc(ND * 2);
  __bf16* f1b  = (__bf16*)alloc(ND * 2);
  float*  f2f  = (float*) alloc(ND * 4);

  // 1) convert activations + weights to bf16
  f32_to_bf16<<<(int)((ND + 255) / 256), 256, 0, stream>>>(x, xb, (int)ND);
  f32_to_bf16<<<(int)((WD + 255) / 256), 256, 0, stream>>>(wq, wqb, (int)WD);
  f32_to_bf16<<<(int)((WD + 255) / 256), 256, 0, stream>>>(wk, wkb, (int)WD);
  f32_to_bf16<<<(int)((WD + 255) / 256), 256, 0, stream>>>(wv, wvb, (int)WD);
  f32_to_bf16<<<(int)((WD + 255) / 256), 256, 0, stream>>>(wo, wob, (int)WD);
  f32_to_bf16<<<(int)((WD + 255) / 256), 256, 0, stream>>>(w1, w1b, (int)WD);
  f32_to_bf16<<<(int)((WD + 255) / 256), 256, 0, stream>>>(w2, w2b, (int)WD);

  const int M = PB * PS;                       // 8192
  const int gemmGrid = (M / 128) * (PD / 64);  // 512 blocks
  const float qscale = 0.125f;                 // 1/sqrt(64)

  // 2) QKV projections
  gemm_bf16<<<gemmGrid, 256, 0, stream>>>(xb, wqb, bq, nullptr, qb,  M, PD, PD, qscale, MODE_HEADS);
  gemm_bf16<<<gemmGrid, 256, 0, stream>>>(xb, wkb, bk, nullptr, kb,  M, PD, PD, 1.0f,   MODE_HEADS);
  gemm_bf16<<<gemmGrid, 256, 0, stream>>>(xb, wvb, bv, nullptr, vtb, M, PD, PD, 1.0f,   MODE_HEADS_T);

  // 3) attention: B*H*(S/16) = 4096 wave-tasks, 8 waves/block
  flash_attn<<<512, 256, 0, stream>>>(qb, kb, vtb, ctxb);

  // 4) output projection -> sa (f32)
  gemm_bf16<<<gemmGrid, 256, 0, stream>>>(ctxb, wob, bo, sa, nullptr, M, PD, PD, 1.0f, MODE_F32);

  // 5) h = LN(sa + x) -> hf (f32) + hb (bf16)
  ln_residual<<<M, 256, 0, stream>>>(sa, x, g1, be1, hf, hb);

  // 6) FFN: f1 = gelu(h@w1 + b1) (bf16), f2 = f1@w2 + b2 (f32)
  gemm_bf16<<<gemmGrid, 256, 0, stream>>>(hb, w1b, b1, nullptr, f1b, M, PD, PD, 1.0f, MODE_GELU);
  gemm_bf16<<<gemmGrid, 256, 0, stream>>>(f1b, w2b, b2, f2f, nullptr, M, PD, PD, 1.0f, MODE_F32);

  // 7) out = LN(f2 + h)
  ln_residual<<<M, 256, 0, stream>>>(f2f, hf, g2, be2, out, nullptr);
}
